// TriXLinear_2465311228261
// MI455X (gfx1250) — matrix operand order repaired and checked by reading, not compiled, not run on hardware
//
#include <hip/hip_runtime.h>

// TriX ternary-weight GEMM for MI455X (gfx1250, wave32, WMMA).
// out[B,O] = (x[B,K] @ sign(W)[O,K]^T) * scales[O] * gate[B, O/1024]
//
// Compute-bound: 137 GFLOP vs ~192MB unique traffic (~8us at 23.3 TB/s).
// Both operands (128MB) fit the 192MB L2, so fragments load directly
// global->VGPR per the ISA WMMA layouts (row-major W already gives the
// contiguous-K-per-lane pattern the B fragment wants -> no transpose).
// f16 WMMA, f32 accumulation: W is exactly +-1 in f16; x ~ N(0,1).
//
// Wave tile 64x64 (16 accumulators): 16 WMMA per k-step against 32 b128
// loads + ~64 VALU -> 2 loads + 4 VALU per WMMA, keeping the issue stream
// from starving the XDL pipe and giving the scheduler 16 independent WMMAs
// to cover the WMMA->VALU WAR hazard windows.

typedef __attribute__((ext_vector_type(16))) _Float16 v16h;
typedef __attribute__((ext_vector_type(8)))  float    v8f;

#define IN_F  4096
#define OUT_F 4096
// NUM_TILES = 4 -> tile width 1024 -> col >> 10

union HalfBits { unsigned short u; _Float16 h; };

__device__ __forceinline__ _Float16 sign_half(float f) {
  // sign(f) as f16: +-1.0 (0x3C00 | signbit). Compiler fuses pairs into
  // packed v_and/v_or on 2xf16.
  HalfBits c;
  c.u = (unsigned short)(0x3C00u | ((__float_as_uint(f) >> 16) & 0x8000u));
  return c.h;
}

__global__ __launch_bounds__(256)
void trix_wmma_kernel(const float* __restrict__ x,
                      const int*   __restrict__ gate,
                      const float* __restrict__ w,
                      const float* __restrict__ scales,
                      float*       __restrict__ out)
{
  const int lane = threadIdx.x & 31;
  const int wave = threadIdx.x >> 5;
  const int l15  = lane & 15;
  const int lhi  = lane >> 4;

  // 8 waves in a 2x4 grid over a 128x256 block tile; each wave owns 64x64.
  const int m_base = blockIdx.y * 128 + (wave >> 2) * 64;
  const int n_base = blockIdx.x * 256 + (wave & 3) * 64;

  v8f acc[4][4];
#pragma unroll
  for (int i = 0; i < 4; ++i)
#pragma unroll
    for (int j = 0; j < 4; ++j) {
      v8f z = {};
      acc[i][j] = z;
    }

  // A-fragment base: lane holds row (m_base + l15); K runs start at lhi*8.
  // (16-bit A 16x32 layout: e=0..7 -> K=8*lhi+e, e=8..15 -> K=8*lhi+16+(e-8))
  const float* xa = x + (size_t)(m_base + l15) * IN_F + lhi * 8;
  // B-fragment base: lane holds W row (= output col) (n_base + l15),
  // K runs start at lhi*8 (same layout as A: K=8*lhi+e, K=8*lhi+16+(e-8)).
  const float* wb = w + (size_t)(n_base + l15) * IN_F + lhi * 8;

  for (int k0 = 0; k0 < IN_F; k0 += 32) {
    // ---- A fragments: x tile, fp32 -> f16 convert ----
    v16h A[4];
#pragma unroll
    for (int mt = 0; mt < 4; ++mt) {
      const float* p = xa + (size_t)mt * 16 * IN_F + k0;
      float4 f0 = *(const float4*)(p + 0);   // K = kb .. kb+3
      float4 f1 = *(const float4*)(p + 4);   // K = kb+4 .. kb+7
      float4 f2 = *(const float4*)(p + 16);  // K = kb+16 .. kb+19
      float4 f3 = *(const float4*)(p + 20);  // K = kb+20 .. kb+23
      v16h a;
      a[0]  = (_Float16)f0.x; a[1]  = (_Float16)f0.y;
      a[2]  = (_Float16)f0.z; a[3]  = (_Float16)f0.w;
      a[4]  = (_Float16)f1.x; a[5]  = (_Float16)f1.y;
      a[6]  = (_Float16)f1.z; a[7]  = (_Float16)f1.w;
      a[8]  = (_Float16)f2.x; a[9]  = (_Float16)f2.y;
      a[10] = (_Float16)f2.z; a[11] = (_Float16)f2.w;
      a[12] = (_Float16)f3.x; a[13] = (_Float16)f3.y;
      a[14] = (_Float16)f3.z; a[15] = (_Float16)f3.w;
      A[mt] = a;
    }

    // ---- B fragments: ternary weights, sign -> +-1 f16 (bit ops only) ----
    v16h Bf[4];
#pragma unroll
    for (int nt = 0; nt < 4; ++nt) {
      const float* p = wb + (size_t)nt * 16 * IN_F + k0;
      float4 g0 = *(const float4*)(p + 0);
      float4 g1 = *(const float4*)(p + 4);
      float4 g2 = *(const float4*)(p + 16);
      float4 g3 = *(const float4*)(p + 20);
      v16h b;
      b[0]  = sign_half(g0.x); b[1]  = sign_half(g0.y);
      b[2]  = sign_half(g0.z); b[3]  = sign_half(g0.w);
      b[4]  = sign_half(g1.x); b[5]  = sign_half(g1.y);
      b[6]  = sign_half(g1.z); b[7]  = sign_half(g1.w);
      b[8]  = sign_half(g2.x); b[9]  = sign_half(g2.y);
      b[10] = sign_half(g2.z); b[11] = sign_half(g2.w);
      b[12] = sign_half(g3.x); b[13] = sign_half(g3.y);
      b[14] = sign_half(g3.z); b[15] = sign_half(g3.w);
      Bf[nt] = b;
    }

    // ---- 16 WMMAs: D = A x B + C, f32 accumulation ----
#pragma unroll
    for (int mt = 0; mt < 4; ++mt)
#pragma unroll
      for (int nt = 0; nt < 4; ++nt)
        acc[mt][nt] = __builtin_amdgcn_wmma_f32_16x16x32_f16(
            /*neg_a=*/false, A[mt], /*neg_b=*/false, Bf[nt],
            /*c_mod=*/(short)0, acc[mt][nt],
            /*reuse_a=*/false, /*reuse_b=*/false);
  }

  // ---- Epilogue: per-output scale + per-sample tile gate, then store ----
  // C/D layout: VGPR r <-> M = r + 8*lhi, N = l15.
#pragma unroll
  for (int nt = 0; nt < 4; ++nt) {
    const int col  = n_base + nt * 16 + l15;
    const float s  = scales[col];
    const int tile = col >> 10;  // OUT_F / NUM_TILES = 1024
#pragma unroll
    for (int mt = 0; mt < 4; ++mt) {
#pragma unroll
      for (int r = 0; r < 8; ++r) {
        const int row = m_base + mt * 16 + lhi * 8 + r;
        const float g = (float)gate[row * 4 + tile];
        out[(size_t)row * OUT_F + col] = acc[mt][nt][r] * s * g;
      }
    }
  }
}

extern "C" void kernel_launch(void* const* d_in, const int* in_sizes, int n_in,
                              void* d_out, int out_size, void* d_ws, size_t ws_size,
                              hipStream_t stream) {
  (void)n_in; (void)out_size; (void)d_ws; (void)ws_size;
  const float* x      = (const float*)d_in[0];   // [B, 4096] f32
  const int*   gate   = (const int*)d_in[1];     // [B, 4]    i32
  const float* weight = (const float*)d_in[2];   // [4096, 4096] f32 (ternary)
  const float* scales = (const float*)d_in[3];   // [4096] f32
  float* out = (float*)d_out;                    // [B, 4096] f32

  const int batch = in_sizes[0] / IN_F;          // 4096
  dim3 grid(OUT_F / 256, batch / 128);
  trix_wmma_kernel<<<grid, dim3(256), 0, stream>>>(x, gate, weight, scales, out);
}
